// GRU_19129784337067
// MI455X (gfx1250) — matrix operand order repaired and checked by reading, not compiled, not run on hardware
//
#include <hip/hip_runtime.h>
#include <stdint.h>

// ---------------- problem dims ----------------
#define T_STEPS 2048
#define B_DIM   64
#define F_DIM   256
#define H_DIM   256
#define NH3     768
#define BH      (B_DIM * H_DIM)

#define NBLK    8          // persistent blocks
#define NTHR    256        // 8 waves of 32
#define W_MAT_STRIDE (48 * 8 * 32 * 8)   // 98304 dwords per swizzled matrix (384 KB)
#define LDS_DWORDS (12 * 8 * 32 * 8)     // 24576 dwords = 96 KB (12 swizzled tiles)

typedef __bf16        v16bf __attribute__((ext_vector_type(16)));
typedef __bf16        v2bf  __attribute__((ext_vector_type(2)));
typedef float         v8f   __attribute__((ext_vector_type(8)));
typedef unsigned int  v8u   __attribute__((ext_vector_type(8)));

// ---------------- helpers ----------------
__device__ __forceinline__ unsigned bf16_rne(float f) {
    unsigned u = __builtin_bit_cast(unsigned, f);
    return (u + 0x7FFFu + ((u >> 16) & 1u)) >> 16;
}

// pack two f32 -> packed bf16x2 dword (lo in [15:0], hi in [31:16])
__device__ __forceinline__ unsigned pk_bf16x2(float lo, float hi) {
#if __has_builtin(__builtin_amdgcn_cvt_pk_bf16_f32)
    v2bf p = __builtin_amdgcn_cvt_pk_bf16_f32(lo, hi);
    return __builtin_bit_cast(unsigned, p);
#else
    unsigned a = __builtin_bit_cast(unsigned, lo);
    unsigned b = __builtin_bit_cast(unsigned, hi);
    return (a >> 16) | (b & 0xFFFF0000u);
#endif
}

__device__ __forceinline__ float sigm_f(float x) {
    return 1.0f / (1.0f + __expf(-x));
}
__device__ __forceinline__ float tanh_f(float x) {
    x = fminf(fmaxf(x, -15.0f), 15.0f);
    float e = __expf(2.0f * x);
    return (e - 1.0f) / (e + 1.0f);
}

// Load 16x32 f32 A-tile chunk (rows m0..m0+15, K = kc*32..+31) from row-major
// memory, converted to bf16 in the CDNA5 WMMA 16-bit A layout:
//   lane l<16 : M=l,    K = {0..7, 16..23}  of the chunk
//   lane l>=16: M=l-16, K = {8..15, 24..31} of the chunk
__device__ __forceinline__ v16bf load_a_bf16(const float* __restrict__ base,
                                             int rowStride, int m0, int kc,
                                             int lm, int lhi) {
    const float* p = base + (m0 + lm) * rowStride + kc * 32 + lhi * 8;
    float4 f0 = *(const float4*)(p + 0);
    float4 f1 = *(const float4*)(p + 4);
    float4 f2 = *(const float4*)(p + 16);
    float4 f3 = *(const float4*)(p + 20);
    v8u u;
    u[0] = pk_bf16x2(f0.x, f0.y); u[1] = pk_bf16x2(f0.z, f0.w);
    u[2] = pk_bf16x2(f1.x, f1.y); u[3] = pk_bf16x2(f1.z, f1.w);
    u[4] = pk_bf16x2(f2.x, f2.y); u[5] = pk_bf16x2(f2.z, f2.w);
    u[6] = pk_bf16x2(f3.x, f3.y); u[7] = pk_bf16x2(f3.z, f3.w);
    return __builtin_bit_cast(v16bf, u);
}

// Pre-swizzled B tile chunk from LDS
__device__ __forceinline__ v16bf load_b_lds(const unsigned* lds,
                                            int tl, int kc, int lane) {
    const unsigned* p = lds + ((tl * 8 + kc) << 8) + lane * 8;
    uint4 a = *(const uint4*)(p);
    uint4 b = *(const uint4*)(p + 4);
    v8u u;
    u[0] = a.x; u[1] = a.y; u[2] = a.z; u[3] = a.w;
    u[4] = b.x; u[5] = b.y; u[6] = b.z; u[7] = b.w;
    return __builtin_bit_cast(v16bf, u);
}

#define WMMA_BF16(A, Bm, C) \
    __builtin_amdgcn_wmma_f32_16x16x32_bf16(false, (A), false, (Bm), (short)0, (C), false, false)

// ---------------- kernel 0: swizzle Wi/Wh f32 -> bf16 WMMA-B layout ----------------
// ws layout: [mat(2)][t48(48)][kc(8)][lane(32)][j(8)] packed bf16x2 dwords.
// Element (k, n) of W:  lane<16: n=t48*16+lane,     k=kc*32 + (j>>2)*16 + 2(j&3)(+1)
//                       lane>=16: n=t48*16+lane-16, k=kc*32+8 + (j>>2)*16 + 2(j&3)(+1)
__global__ void gru_swizzle_w_kernel(const float* __restrict__ Wi,
                                     const float* __restrict__ Wh,
                                     unsigned* __restrict__ wsW) {
    int idx  = blockIdx.x * 256 + threadIdx.x;      // 0 .. 2*W_MAT_STRIDE-1
    int mat  = idx / W_MAT_STRIDE;
    int r    = idx % W_MAT_STRIDE;
    int t48  = r >> 11;
    int kc   = (r >> 8) & 7;
    int lane = (r >> 3) & 31;
    int j    = r & 7;
    int n    = t48 * 16 + (lane & 15);
    int k    = kc * 32 + ((lane >> 4) << 3) + ((j >> 2) << 4) + 2 * (j & 3);
    const float* src = mat ? Wh : Wi;
    float lo = src[(size_t)k * NH3 + n];
    float hi = src[(size_t)(k + 1) * NH3 + n];
    wsW[idx] = bf16_rne(lo) | (bf16_rne(hi) << 16);
}

// ---------------- kernel 1: persistent fused GRU scan ----------------
// 8 blocks x 8 waves; wave w of block b owns output tile
//   rows  m0 = (w&3)*16   (batch),  cols n0 = (2b + w>>2)*16  (hidden)
// LDS (96 KB dynamic): tiles 0..5 = Wi (cbL x gate), tiles 6..11 = Wh.
__global__ void __launch_bounds__(NTHR, 1)
gru_scan_kernel(const float* __restrict__ c0,
                const float* __restrict__ xs,
                const float* __restrict__ bh,
                const float* __restrict__ b_in,
                const unsigned* __restrict__ wsW,
                unsigned* __restrict__ cnt,
                float* __restrict__ out) {
    extern __shared__ unsigned ldsW[];   // LDS_DWORDS dwords = 96 KB

    const int tid  = threadIdx.x;
    const int wid  = tid >> 5;
    const int lane = tid & 31;
    const int rb   = wid & 3;
    const int cbL  = wid >> 2;                // 0 or 1
    const int cb   = blockIdx.x * 2 + cbL;    // 0..15
    const int m0   = rb * 16;
    const int n0   = cb * 16;
    const int lm   = lane & 15;
    const int lhi  = lane >> 4;

    // stage this block's Wi and Wh swizzled tiles into LDS (once)
    for (int i = tid; i < LDS_DWORDS; i += NTHR) {
        int tl  = i >> 11;               // 0..11
        int rem = i & 2047;
        int mat = (tl >= 6) ? 1 : 0;
        int tlm = tl - mat * 6;
        int g   = tlm % 3;
        int cl  = tlm / 3;
        int t48 = g * 16 + blockIdx.x * 2 + cl;
        ldsW[i] = wsW[(size_t)mat * W_MAT_STRIDE + (t48 << 11) + rem];
    }
    __syncthreads();

    const int tlWi = cbL * 3;        // LDS tile base for Wi gates r/z/n
    const int tlWh = 6 + cbL * 3;    // LDS tile base for Wh gates r/z/n

    // per-lane gate biases (constant over time)
    const int   n   = n0 + lm;
    const float bR  = bh[n];
    const float bZ  = bh[H_DIM + n];
    const float bN  = bh[2 * H_DIM + n];
    const float bI  = b_in[n];

    float* ys = out + BH;
    unsigned target = 0;

    for (int t = 0; t < T_STEPS; ++t) {
        const float* hprev = (t == 0) ? c0 : (ys + (size_t)(t - 1) * BH);
        const float* xt    = xs + (size_t)t * (B_DIM * F_DIM);

        // hoist per-element h_prev loads so their latency hides under the GEMM
        float hp[8];
        #pragma unroll
        for (int v = 0; v < 8; ++v) {
            hp[v] = hprev[(m0 + v + 8 * lhi) * H_DIM + n];
        }

        v8f accR = {}, accZ = {}, accXN = {}, accHN = {};

        #pragma unroll 2
        for (int kc = 0; kc < 8; ++kc) {
            // all loads of this iteration first (A from global, B from LDS) ...
            v16bf ax   = load_a_bf16(xt,    F_DIM, m0, kc, lm, lhi);
            v16bf ah   = load_a_bf16(hprev, H_DIM, m0, kc, lm, lhi);
            v16bf bWiR = load_b_lds(ldsW, tlWi + 0, kc, lane);
            v16bf bWhR = load_b_lds(ldsW, tlWh + 0, kc, lane);
            v16bf bWiZ = load_b_lds(ldsW, tlWi + 1, kc, lane);
            v16bf bWhZ = load_b_lds(ldsW, tlWh + 1, kc, lane);
            v16bf bWiN = load_b_lds(ldsW, tlWi + 2, kc, lane);
            v16bf bWhN = load_b_lds(ldsW, tlWh + 2, kc, lane);

            // ... then the 6 matrix ops
            accR  = WMMA_BF16(ax, bWiR, accR);
            accR  = WMMA_BF16(ah, bWhR, accR);
            accZ  = WMMA_BF16(ax, bWiZ, accZ);
            accZ  = WMMA_BF16(ah, bWhZ, accZ);
            accXN = WMMA_BF16(ax, bWiN, accXN);
            accHN = WMMA_BF16(ah, bWhN, accHN);
        }

        // prefetch next timestep's xs rows into cache (global_prefetch_b8)
        if (t + 1 < T_STEPS) {
            __builtin_prefetch(xt + B_DIM * F_DIM + (m0 + lm) * F_DIM + lhi * 32, 0, 1);
        }

        // gate math + h_new write.  C layout: c[v] is (m = m0+v+8*lhi, n)
        float* yt = ys + (size_t)t * BH;
        const bool last = (t == T_STEPS - 1);
        #pragma unroll
        for (int v = 0; v < 8; ++v) {
            int   off  = (m0 + v + 8 * lhi) * H_DIM + n;
            float r    = sigm_f(accR[v] + bR);
            float z    = sigm_f(accZ[v] + bZ);
            float hn   = accHN[v] + bN;
            float nn   = tanh_f(accXN[v] + bI + r * hn);
            float hnew = nn + z * (hp[v] - nn);   // == (1-z)*nn + z*hp
            yt[off] = hnew;
            if (last) out[off] = hnew;            // final_c
        }

        // ---- grid barrier: publish h(t), wait for all blocks ----
        __threadfence();          // release this thread's stores (agent scope)
        __syncthreads();          // whole block done
        target += NBLK;
        if (tid == 0) {
            __hip_atomic_fetch_add(cnt, 1u, __ATOMIC_RELAXED, __HIP_MEMORY_SCOPE_AGENT);
            while (__hip_atomic_load(cnt, __ATOMIC_RELAXED, __HIP_MEMORY_SCOPE_AGENT) < target) {
                __builtin_amdgcn_s_sleep(1);
            }
        }
        __syncthreads();
        __threadfence();          // acquire: invalidate so we see other blocks' h(t)
    }
}

// ---------------- host launch ----------------
extern "C" void kernel_launch(void* const* d_in, const int* in_sizes, int n_in,
                              void* d_out, int out_size, void* d_ws, size_t ws_size,
                              hipStream_t stream) {
    const float* c0  = (const float*)d_in[0];
    const float* xs  = (const float*)d_in[1];
    const float* Wi  = (const float*)d_in[2];
    const float* Wh  = (const float*)d_in[3];
    const float* bh  = (const float*)d_in[4];
    const float* bin = (const float*)d_in[5];
    float*       out = (float*)d_out;

    unsigned* wsW = (unsigned*)d_ws;                 // 2*W_MAT_STRIDE dwords (768 KB)
    unsigned* cnt = wsW + 2 * W_MAT_STRIDE;          // barrier counter after weights

    (void)in_sizes; (void)n_in; (void)out_size; (void)ws_size;

    hipMemsetAsync(cnt, 0, 256, stream);
    gru_swizzle_w_kernel<<<(2 * W_MAT_STRIDE) / 256, 256, 0, stream>>>(Wi, Wh, wsW);
    gru_scan_kernel<<<NBLK, NTHR, LDS_DWORDS * sizeof(unsigned), stream>>>(
        c0, xs, bh, bin, wsW, cnt, out);
}
